// PremapJoiner_71829033058737
// MI455X (gfx1250) — matrix operand order repaired and checked by reading, not compiled, not run on hardware
//
#include <hip/hip_runtime.h>
#include <hip/hip_bf16.h>
#include <math.h>

typedef __attribute__((ext_vector_type(16))) __bf16 v16bf;
typedef __attribute__((ext_vector_type(8)))  float  v8f;

#define B_  4
#define T_  512
#define U_  128
#define H_  512
#define V_  41
#define NT_ 3                 // 3 N-tiles of 16 cover 48 >= 41 (padded)
#define KSTEPS_ (H_ / 32)     // 16 k-steps of 32 (bf16 WMMA K)

// Branch-free tanh. gfx1250 has native v_tanh_f32; otherwise use the
// hardware exp2/rcp transcendentals: tanh(x) = 1 - 2/(exp2(2x*log2e)+1).
// exp2 -> +inf saturates to +1, exp2 -> 0 saturates to -1. Result is
// rounded to bf16 immediately, so this is bit-adequate precision.
__device__ __forceinline__ float fast_tanh(float x) {
#if __has_builtin(__builtin_amdgcn_tanhf)
    return __builtin_amdgcn_tanhf(x);
#else
    float e = __builtin_amdgcn_exp2f(x * 2.8853900817779268f); // 2*log2(e)
    return 1.0f - 2.0f * __builtin_amdgcn_rcpf(e + 1.0f);
#endif
}

// ---------------------------------------------------------------------------
// Workspace layout:
//   [0, 49152)            : Wl swizzled to WMMA B-fragment order, bf16
//                           idx = ((ks*3 + nt)*32 + lane)*16 + e
//   [49152, 49152 + 1MB)  : tgt_p fp32 [B*U*H]  (premap1 writes GELU(h1),
//                           premap2 rewrites in place with LN(h2))
// ---------------------------------------------------------------------------

__global__ void __launch_bounds__(256)
wl_swizzle_kernel(const float* __restrict__ Wl, __bf16* __restrict__ wlsw) {
    int idx = blockIdx.x * blockDim.x + threadIdx.x;
    const int total = KSTEPS_ * NT_ * 32 * 16;
    if (idx >= total) return;
    int e    = idx & 15;
    int lane = (idx >> 4) & 31;
    int nt   = (idx >> 9) % NT_;
    int ks   = idx / (512 * NT_);
    // B-matrix (32x16 bf16) layout: lanes 0-15 hold K=0..7,16..23 (2/VGPR),
    // lanes 16-31 hold K=8..15,24..31; N = lane%16.
    int K = ks * 32 + ((e >> 3) << 4) + ((lane >> 4) << 3) + (e & 7);
    int N = nt * 16 + (lane & 15);
    float v = (N < V_) ? Wl[(size_t)K * V_ + N] : 0.0f;
    wlsw[idx] = (__bf16)v;
}

// h1 = GELU_exact(tgt @ W1 + b1) -> buf   (one block per row of B*U)
__global__ void __launch_bounds__(256)
premap1_kernel(const float* __restrict__ tgt, const float* __restrict__ W1,
               const float* __restrict__ b1, float* __restrict__ buf) {
    __shared__ float row[H_];
    int r   = blockIdx.x;
    int tid = threadIdx.x;
    const float* trow = tgt + (size_t)r * H_;
    row[tid]       = trow[tid];
    row[tid + 256] = trow[tid + 256];
    __syncthreads();
    #pragma unroll
    for (int cc = 0; cc < 2; ++cc) {
        int c = tid + cc * 256;
        float acc = b1[c];
        for (int h = 0; h < H_; ++h)
            acc = fmaf(row[h], W1[(size_t)h * H_ + c], acc);
        float g = 0.5f * acc * (1.0f + erff(acc * 0.7071067811865476f));
        buf[(size_t)r * H_ + c] = g;
    }
}

// tgt_p = LayerNorm(buf @ W2 + b2) * ln_g + ln_b  (in place on buf)
__global__ void __launch_bounds__(256)
premap2_kernel(float* __restrict__ buf, const float* __restrict__ W2,
               const float* __restrict__ b2, const float* __restrict__ ln_g,
               const float* __restrict__ ln_b) {
    __shared__ float row[H_];
    __shared__ float hbuf[H_];
    __shared__ float rs[256], rq[256];
    int r   = blockIdx.x;
    int tid = threadIdx.x;
    float* brow = buf + (size_t)r * H_;
    row[tid]       = brow[tid];
    row[tid + 256] = brow[tid + 256];
    __syncthreads();
    float v0, v1;
    {
        float acc = b2[tid];
        for (int h = 0; h < H_; ++h)
            acc = fmaf(row[h], W2[(size_t)h * H_ + tid], acc);
        v0 = acc; hbuf[tid] = acc;
    }
    {
        int c = tid + 256;
        float acc = b2[c];
        for (int h = 0; h < H_; ++h)
            acc = fmaf(row[h], W2[(size_t)h * H_ + c], acc);
        v1 = acc; hbuf[c] = acc;
    }
    rs[tid] = v0 + v1;
    rq[tid] = v0 * v0 + v1 * v1;
    __syncthreads();
    for (int s = 128; s > 0; s >>= 1) {
        if (tid < s) { rs[tid] += rs[tid + s]; rq[tid] += rq[tid + s]; }
        __syncthreads();
    }
    float mean = rs[0] * (1.0f / H_);
    float var  = rq[0] * (1.0f / H_) - mean * mean;
    float inv  = rsqrtf(var + 1e-5f);
    brow[tid]       = (hbuf[tid]       - mean) * inv * ln_g[tid]       + ln_b[tid];
    brow[tid + 256] = (hbuf[tid + 256] - mean) * inv * ln_g[tid + 256] + ln_b[tid + 256];
}

// Main joiner: out[b,t,u,:] = tanh(src[b,t,:] + tgt_p[b,u,:]) @ Wl + bl
// One block per (b,t); wave w handles u-tile [16w, 16w+16).
__global__ void __launch_bounds__(256)
joiner_main_kernel(const float* __restrict__ src, const float* __restrict__ tgtp,
                   const __bf16* __restrict__ wlsw, const float* __restrict__ bl,
                   float* __restrict__ out) {
    int blk  = blockIdx.x;            // b*T + t
    int b    = blk / T_;
    int tid  = threadIdx.x;
    int wave = tid >> 5;
    int lane = tid & 31;
    int hf   = lane >> 4;             // lane half selects K sub-range
    int mrow = lane & 15;             // A-matrix row (M) within tile
    int u0   = wave * 16;

    const float* srow = src  + (size_t)blk * H_;
    const float* grow = tgtp + ((size_t)(b * U_ + u0 + mrow)) * H_;
    const v16bf* bbase = (const v16bf*)wlsw;

    v8f acc0 = {}, acc1 = {}, acc2 = {};

    for (int ks = 0; ks < KSTEPS_; ++ks) {
        int kbase = ks * 32;
        // bf16 A fragment, ISA 16-bit A 16x32 layout. For fixed lane-half and
        // K-group g, the 8 element K's are contiguous: K = kbase+16g+8hf+j.
        v16bf a;
        #pragma unroll
        for (int g = 0; g < 2; ++g) {
            int kb = kbase + (g << 4) + (hf << 3);
            float4 s0 = *(const float4*)(srow + kb);
            float4 s1 = *(const float4*)(srow + kb + 4);
            float4 t0 = *(const float4*)(grow + kb);
            float4 t1 = *(const float4*)(grow + kb + 4);
            a[8 * g + 0] = (__bf16)fast_tanh(s0.x + t0.x);
            a[8 * g + 1] = (__bf16)fast_tanh(s0.y + t0.y);
            a[8 * g + 2] = (__bf16)fast_tanh(s0.z + t0.z);
            a[8 * g + 3] = (__bf16)fast_tanh(s0.w + t0.w);
            a[8 * g + 4] = (__bf16)fast_tanh(s1.x + t1.x);
            a[8 * g + 5] = (__bf16)fast_tanh(s1.y + t1.y);
            a[8 * g + 6] = (__bf16)fast_tanh(s1.z + t1.z);
            a[8 * g + 7] = (__bf16)fast_tanh(s1.w + t1.w);
        }
        v16bf bf0 = bbase[(ks * NT_ + 0) * 32 + lane];
        v16bf bf1 = bbase[(ks * NT_ + 1) * 32 + lane];
        v16bf bf2 = bbase[(ks * NT_ + 2) * 32 + lane];
        acc0 = __builtin_amdgcn_wmma_f32_16x16x32_bf16(false, a, false, bf0,
                                                       (short)0, acc0, false, false);
        acc1 = __builtin_amdgcn_wmma_f32_16x16x32_bf16(false, a, false, bf1,
                                                       (short)0, acc1, false, false);
        acc2 = __builtin_amdgcn_wmma_f32_16x16x32_bf16(false, a, false, bf2,
                                                       (short)0, acc2, false, false);
    }

    // C/D layout: VGPR i, lane L -> M = i + 8*(L/16), N = L%16.
    float bl0 = bl[mrow];
    float bl1 = bl[16 + mrow];
    float bl2 = (32 + mrow < V_) ? bl[32 + mrow] : 0.0f;
    #pragma unroll
    for (int i = 0; i < 8; ++i) {
        int u = u0 + i + (hf << 3);
        size_t ro = ((size_t)blk * U_ + u) * V_;
        out[ro + mrow]      = acc0[i] + bl0;
        out[ro + 16 + mrow] = acc1[i] + bl1;
        if (32 + mrow < V_)
            out[ro + 32 + mrow] = acc2[i] + bl2;
    }
}

extern "C" void kernel_launch(void* const* d_in, const int* in_sizes, int n_in,
                              void* d_out, int out_size, void* d_ws, size_t ws_size,
                              hipStream_t stream) {
    (void)in_sizes; (void)n_in; (void)out_size; (void)ws_size;
    const float* src  = (const float*)d_in[0];
    const int*   slen = (const int*)d_in[1];
    const float* tgt  = (const float*)d_in[2];
    const int*   tlen = (const int*)d_in[3];
    const float* W1   = (const float*)d_in[4];
    const float* b1   = (const float*)d_in[5];
    const float* W2   = (const float*)d_in[6];
    const float* b2   = (const float*)d_in[7];
    const float* lng  = (const float*)d_in[8];
    const float* lnb  = (const float*)d_in[9];
    const float* Wl   = (const float*)d_in[10];
    const float* bl   = (const float*)d_in[11];
    float* out = (float*)d_out;
    char*  ws  = (char*)d_ws;

    __bf16* wlsw = (__bf16*)(ws);                       // 49152 bytes
    float*  buf  = (float*)(ws + 49152);                // B*U*H fp32 (1 MB)

    wl_swizzle_kernel<<<(KSTEPS_ * NT_ * 32 * 16 + 255) / 256, 256, 0, stream>>>(Wl, wlsw);
    premap1_kernel<<<B_ * U_, 256, 0, stream>>>(tgt, W1, b1, buf);
    premap2_kernel<<<B_ * U_, 256, 0, stream>>>(buf, W2, b2, lng, lnb);
    joiner_main_kernel<<<B_ * T_, 256, 0, stream>>>(src, buf, wlsw, bl, out);

    // Tuple tail: source_lengths then target_lengths, bit-preserved.
    float* tail = out + (size_t)B_ * T_ * U_ * V_;
    hipMemcpyAsync((void*)tail,        (const void*)slen, B_ * sizeof(int),
                   hipMemcpyDeviceToDevice, stream);
    hipMemcpyAsync((void*)(tail + B_), (const void*)tlen, B_ * sizeof(int),
                   hipMemcpyDeviceToDevice, stream);
}
